// GNNVAEModel_34574486733476
// MI455X (gfx1250) — matrix operand order repaired and checked by reading, not compiled, not run on hardware
//
#include <hip/hip_runtime.h>
#include <hip/hip_bf16.h>
#include <math.h>

#define BN_EPS 1e-5f

typedef __attribute__((ext_vector_type(16))) _Float16 v16h;
typedef __attribute__((ext_vector_type(8)))  _Float16 v8h;
typedef __attribute__((ext_vector_type(8)))  float    v8f;
typedef __attribute__((ext_vector_type(4)))  int      v4i;

#if defined(__gfx1250__) && __has_builtin(__builtin_amdgcn_global_load_async_to_lds_b128)
#define HAVE_ASYNC_COPY 1
#else
#define HAVE_ASYNC_COPY 0
#endif

#if HAVE_ASYNC_COPY
__device__ __forceinline__ void wait_async0() {
#if __has_builtin(__builtin_amdgcn_s_wait_asynccnt)
  __builtin_amdgcn_s_wait_asynccnt(0);
#else
  asm volatile("s_wait_asynccnt 0" ::: "memory");
#endif
}
__device__ __forceinline__ void async_copy_b128(const float* gsrc, float* lds) {
  __builtin_amdgcn_global_load_async_to_lds_b128(
      (__attribute__((address_space(1))) v4i*)gsrc,
      (__attribute__((address_space(3))) v4i*)lds,
      0, 0);
}
#endif

// ---------------------------------------------------------------------------
// Tiled WMMA GEMM:  C[M,N] = A[M,K] @ B[K,N] (+ bias[N] if bias != nullptr)
// Block tile 256(M) x 64(N), 8 waves (256 thr), K-chunks of 32.
// Each wave computes a 32x64 strip: 2 A-frags x 4 B-frags = 8 WMMAs/chunk.
// LDS layouts chosen so a fragment is two aligned 16B ds_load_b128:
//   sA row-major, row stride 32 halves (lane's halves = two 8-half runs)
//   sB column-major, column stride 40 halves (lane's halves = two 8-half runs)
// ---------------------------------------------------------------------------
#define GEMM_BM 256
#define GEMM_BN 64
#define GEMM_TK 32
#define B_STRIDE 40   // halves per column (32 + 8 pad), keeps 16B alignment

__global__ __launch_bounds__(256) void gemm_wmma_f16(
    const float* __restrict__ A, const float* __restrict__ B,
    const float* __restrict__ bias, float* __restrict__ C,
    int M, int N, int K)
{
  __shared__ _Float16 sA[GEMM_BM * GEMM_TK];          // 16 KB
  __shared__ _Float16 sB[GEMM_BN * B_STRIDE];         // 5 KB, column-major
#if HAVE_ASYNC_COPY
  __shared__ float    sBraw[GEMM_TK * GEMM_BN];       // 8 KB raw f32 staging
#endif

  const int tid  = threadIdx.x;
  const int wave = tid >> 5;
  const int lane = tid & 31;
  const int hl   = lane & 15;   // half-lane (0..15)
  const int hi   = lane >> 4;   // lane group (0 or 1)
  const int m0   = blockIdx.y * GEMM_BM;
  const int n0   = blockIdx.x * GEMM_BN;

  v8f acc[2][4] = {};           // two 16-row frags x four 16-col frags

  const int nChunks = (K + GEMM_TK - 1) / GEMM_TK;
  const bool kEven = (K % 2 == 0);

  for (int ck = 0; ck < nChunks; ++ck) {
    const int kb = ck * GEMM_TK;

    // ---- Stage A tile: 256 x 32 row-major, f32 -> f16 (float2 fast path) ---
    for (int s = tid; s < (GEMM_BM * GEMM_TK) / 2; s += 256) {
      int r  = s >> 4;                 // 16 float2-pairs per row
      int kk = (s & 15) * 2;
      int gr = m0 + r, gk = kb + kk;
      _Float16 h0, h1;
      if (kEven && gr < M && gk + 1 < K) {
        float2 v = *(const float2*)(A + (size_t)gr * K + gk);
        h0 = (_Float16)v.x; h1 = (_Float16)v.y;
      } else {
        float a0 = (gr < M && gk     < K) ? A[(size_t)gr * K + gk]     : 0.0f;
        float a1 = (gr < M && gk + 1 < K) ? A[(size_t)gr * K + gk + 1] : 0.0f;
        h0 = (_Float16)a0; h1 = (_Float16)a1;
      }
      *(__attribute__((ext_vector_type(2))) _Float16*)&sA[r * GEMM_TK + kk] =
          (__attribute__((ext_vector_type(2))) _Float16){h0, h1};
    }

    // ---- Stage B tile: 32 x 64 -> column-major f16 ------------------------
    const bool fullB = (kb + GEMM_TK <= K) && (n0 + GEMM_BN <= N);
#if HAVE_ASYNC_COPY
    if (fullB) {
      // CDNA5 async DMA: raw f32 tile -> LDS (ASYNCcnt), then convert in LDS.
      for (int s4 = tid; s4 < (GEMM_TK * GEMM_BN) / 4; s4 += 256) {
        int kk = s4 >> 4;              // 16 float4 per k-row
        int c  = (s4 & 15) * 4;
        async_copy_b128(B + (size_t)(kb + kk) * N + n0 + c,
                        &sBraw[kk * GEMM_BN + c]);
      }
      wait_async0();
      __syncthreads();   // all async writes visible before conversion
      for (int idx = tid; idx < GEMM_TK * GEMM_BN; idx += 256) {
        int kk = idx >> 6, c = idx & 63;
        sB[c * B_STRIDE + kk] = (_Float16)sBraw[idx];
      }
    } else
#endif
    {
      for (int idx = tid; idx < GEMM_TK * GEMM_BN; idx += 256) {
        int kk = idx >> 6, c = idx & 63;
        int gk = kb + kk, gc = n0 + c;
        float v = (gk < K && gc < N) ? B[(size_t)gk * N + gc] : 0.0f;
        sB[c * B_STRIDE + kk] = (_Float16)v;
      }
    }

    // CDNA5 speculative prefetch of next A chunk (global_prefetch_b8).
    if (kb + GEMM_TK < K) {
      int pr = m0 + tid;
      if (pr < M) __builtin_prefetch(&A[(size_t)pr * K + kb + GEMM_TK], 0, 1);
    }
    __syncthreads();

    // ---- A fragments: two aligned b128 loads each -------------------------
    // Lane hl holds row; halves = {k = hi*8..+7} ++ {k = 16+hi*8..+7}.
    v16h afr[2];
    #pragma unroll
    for (int f = 0; f < 2; ++f) {
      const v8h* ap = (const v8h*)&sA[(wave * 32 + f * 16 + hl) * GEMM_TK];
      v8h lo = ap[hi];
      v8h hp = ap[2 + hi];
      afr[f] = __builtin_shufflevector(lo, hp, 0,1,2,3,4,5,6,7,8,9,10,11,12,13,14,15);
    }

    #pragma unroll
    for (int t = 0; t < 4; ++t) {
      // B fragment: column n = t*16+hl, halves = {k = hi*8..+7} ++ {k = 16+hi*8..+7}.
      const v8h* bp = (const v8h*)&sB[(t * 16 + hl) * B_STRIDE];
      v8h lo = bp[hi];
      v8h hp = bp[2 + hi];
      v16h bf = __builtin_shufflevector(lo, hp, 0,1,2,3,4,5,6,7,8,9,10,11,12,13,14,15);
      acc[0][t] = __builtin_amdgcn_wmma_f32_16x16x32_f16(
          false, afr[0], false, bf, (short)0, acc[0][t], false, false);
      acc[1][t] = __builtin_amdgcn_wmma_f32_16x16x32_f16(
          false, afr[1], false, bf, (short)0, acc[1][t], false, false);
    }
    __syncthreads();
  }

  // ---- Store D (+bias). VGPR v = row (v + 8*hi), column hl per 16x16 tile.
  #pragma unroll
  for (int t = 0; t < 4; ++t) {
    int nc = n0 + t * 16 + hl;
    if (nc >= N) continue;
    float badd = bias ? bias[nc] : 0.0f;
    #pragma unroll
    for (int f = 0; f < 2; ++f) {
      #pragma unroll
      for (int v = 0; v < 8; ++v) {
        int r = m0 + wave * 32 + f * 16 + v + (hi << 3);
        if (r < M) C[(size_t)r * N + nc] = acc[f][t][v] + badd;
      }
    }
  }
}

// ---------------------------------------------------------------------------
// BatchNorm batch statistics: one block per column, reduce over rows.
// ---------------------------------------------------------------------------
__global__ __launch_bounds__(256) void bn_stats(
    const float* __restrict__ X, float* __restrict__ mean,
    float* __restrict__ var, int Nrows, int C)
{
  __shared__ float s1[256], s2[256];
  int col = blockIdx.x;
  float a = 0.0f, b = 0.0f;
  for (int r = threadIdx.x; r < Nrows; r += 256) {
    float v = X[(size_t)r * C + col];
    a += v; b += v * v;
  }
  s1[threadIdx.x] = a; s2[threadIdx.x] = b;
  __syncthreads();
  for (int s = 128; s > 0; s >>= 1) {
    if (threadIdx.x < s) { s1[threadIdx.x] += s1[threadIdx.x + s];
                           s2[threadIdx.x] += s2[threadIdx.x + s]; }
    __syncthreads();
  }
  if (threadIdx.x == 0) {
    float m = s1[0] / (float)Nrows;
    mean[col] = m;
    var[col]  = s2[0] / (float)Nrows - m * m;  // biased (ddof=0) like jnp.var
  }
}

__global__ __launch_bounds__(256) void bn_elu(
    float* __restrict__ X, const float* __restrict__ mean,
    const float* __restrict__ var, const float* __restrict__ g,
    const float* __restrict__ bt, int Nrows, int C)
{
  size_t tot = (size_t)Nrows * C;
  for (size_t i = (size_t)blockIdx.x * 256 + threadIdx.x; i < tot;
       i += (size_t)gridDim.x * 256) {
    int c = (int)(i % C);
    float y = g[c] * (X[i] - mean[c]) * rsqrtf(var[c] + BN_EPS) + bt[c];
    X[i] = (y > 0.0f) ? y : expm1f(y);
  }
}

// ---------------------------------------------------------------------------
// GAT helpers
// ---------------------------------------------------------------------------
__device__ inline unsigned f2o(float f) {          // monotonic float->uint
  unsigned u = __float_as_uint(f);
  return (u & 0x80000000u) ? ~u : (u | 0x80000000u);
}
__device__ inline float o2f(unsigned o) {
  unsigned u = (o & 0x80000000u) ? (o & 0x7FFFFFFFu) : ~o;
  return __uint_as_float(u);
}
__device__ inline void edge_nodes(const int* __restrict__ ei, int E, int e,
                                  int& s, int& d) {
  if (e < E) { s = ei[e]; d = ei[E + e]; }
  else       { s = e - E; d = e - E; }     // appended self-loops
}

__global__ __launch_bounds__(256) void gat_scores(
    const float* __restrict__ hh, const float* __restrict__ asrc,
    const float* __restrict__ adst, float* __restrict__ ssrc,
    float* __restrict__ sdst, int Nn, int H, int Cc)
{
  int i = blockIdx.x * 256 + threadIdx.x;
  if (i >= Nn * H) return;
  int n = i / H, h = i % H;
  const float* row = hh + (size_t)n * H * Cc + h * Cc;
  const float* as  = asrc + h * Cc;
  const float* ad  = adst + h * Cc;
  float s1 = 0.0f, s2 = 0.0f;
  for (int c = 0; c < Cc; ++c) { float v = row[c]; s1 += v * as[c]; s2 += v * ad[c]; }
  ssrc[i] = s1; sdst[i] = s2;
}

__global__ __launch_bounds__(256) void edge_max(
    const int* __restrict__ ei, int E, int Etot,
    const float* __restrict__ ssrc, const float* __restrict__ sdst,
    float* __restrict__ esc, unsigned* __restrict__ mord, int H)
{
  long total = (long)Etot * H;
  for (long i = (long)blockIdx.x * 256 + threadIdx.x; i < total;
       i += (long)gridDim.x * 256) {
    int e = (int)(i / H), h = (int)(i % H);
    int s, d; edge_nodes(ei, E, e, s, d);
    float x = ssrc[(size_t)s * H + h] + sdst[(size_t)d * H + h];
    x = (x > 0.0f) ? x : 0.2f * x;            // LeakyReLU(0.2)
    esc[i] = x;
    atomicMax(&mord[(size_t)d * H + h], f2o(x));
  }
}

__global__ __launch_bounds__(256) void edge_expsum(
    const int* __restrict__ ei, int E, int Etot,
    float* __restrict__ esc, const unsigned* __restrict__ mord,
    float* __restrict__ den, int H)
{
  long total = (long)Etot * H;
  for (long i = (long)blockIdx.x * 256 + threadIdx.x; i < total;
       i += (long)gridDim.x * 256) {
    int e = (int)(i / H), h = (int)(i % H);
    int s, d; edge_nodes(ei, E, e, s, d);
    float m = o2f(mord[(size_t)d * H + h]);
    float p = expf(esc[i] - m);
    esc[i] = p;
    atomicAdd(&den[(size_t)d * H + h], p);
  }
}

__global__ __launch_bounds__(256) void edge_agg(
    const int* __restrict__ ei, int E, int Etot,
    const float* __restrict__ hh, const float* __restrict__ esc,
    const float* __restrict__ den, float* __restrict__ out, int H, int Cc)
{
  int HC = H * Cc;
  long total = (long)Etot * HC;
  for (long i = (long)blockIdx.x * 256 + threadIdx.x; i < total;
       i += (long)gridDim.x * 256) {
    int e = (int)(i / HC), j = (int)(i % HC);
    int h = j / Cc;
    int s, d; edge_nodes(ei, E, e, s, d);
    float alpha = esc[(size_t)e * H + h] / den[(size_t)d * H + h];
    atomicAdd(&out[(size_t)d * HC + j], hh[(size_t)s * HC + j] * alpha);
  }
}

__global__ __launch_bounds__(256) void bias_act(
    float* __restrict__ X, const float* __restrict__ b, int Nn, int C,
    int relu, float* __restrict__ copy)
{
  size_t tot = (size_t)Nn * C;
  for (size_t i = (size_t)blockIdx.x * 256 + threadIdx.x; i < tot;
       i += (size_t)gridDim.x * 256) {
    int c = (int)(i % C);
    float y = X[i] + b[c];
    if (relu) y = fmaxf(y, 0.0f);
    X[i] = y;
    if (copy) copy[i] = y;
  }
}

__global__ __launch_bounds__(256) void fill_f32(float* p, float v, long n) {
  for (long i = (long)blockIdx.x * 256 + threadIdx.x; i < n;
       i += (long)gridDim.x * 256) p[i] = v;
}
__global__ __launch_bounds__(256) void fill_u32(unsigned* p, unsigned v, long n) {
  for (long i = (long)blockIdx.x * 256 + threadIdx.x; i < n;
       i += (long)gridDim.x * 256) p[i] = v;
}

// Cluster head: logits = c1 @ W[K,NC] + b, rowwise softmax. NC <= 16.
__global__ __launch_bounds__(256) void cluster_head(
    const float* __restrict__ c1, const float* __restrict__ w,
    const float* __restrict__ b, float* __restrict__ pred,
    int Nn, int K, int NC)
{
  int n = blockIdx.x * 256 + threadIdx.x;
  if (n >= Nn) return;
  float logits[16];
  for (int j = 0; j < NC; ++j) logits[j] = b[j];
  const float* row = c1 + (size_t)n * K;
  for (int k = 0; k < K; ++k) {
    float v = row[k];
    const float* wr = w + (size_t)k * NC;
    for (int j = 0; j < NC; ++j) logits[j] += v * wr[j];
  }
  float mx = logits[0];
  for (int j = 1; j < NC; ++j) mx = fmaxf(mx, logits[j]);
  float s = 0.0f;
  for (int j = 0; j < NC; ++j) { logits[j] = expf(logits[j] - mx); s += logits[j]; }
  float inv = 1.0f / s;
  for (int j = 0; j < NC; ++j) pred[(size_t)n * NC + j] = logits[j] * inv;
}

// ---------------------------------------------------------------------------
// Launch orchestration
// ---------------------------------------------------------------------------
extern "C" void kernel_launch(void* const* d_in, const int* in_sizes, int n_in,
                              void* d_out, int out_size, void* d_ws, size_t ws_size,
                              hipStream_t stream) {
  (void)n_in; (void)out_size; (void)ws_size;

  const float* x        = (const float*)d_in[0];
  const int*   ei       = (const int*)d_in[1];
  const float* enc_w1   = (const float*)d_in[2];
  const float* enc_b1   = (const float*)d_in[3];
  const float* enc_g1   = (const float*)d_in[4];
  const float* enc_bt1  = (const float*)d_in[5];
  const float* enc_w2   = (const float*)d_in[6];
  const float* enc_b2   = (const float*)d_in[7];
  const float* enc_g2   = (const float*)d_in[8];
  const float* enc_bt2  = (const float*)d_in[9];
  const float* gat1_w   = (const float*)d_in[10];
  const float* gat1_as  = (const float*)d_in[11];
  const float* gat1_ad  = (const float*)d_in[12];
  const float* gat1_b   = (const float*)d_in[13];
  const float* gat2_w   = (const float*)d_in[14];
  const float* gat2_as  = (const float*)d_in[15];
  const float* gat2_ad  = (const float*)d_in[16];
  const float* gat2_b   = (const float*)d_in[17];
  const float* gat3_w   = (const float*)d_in[18];
  const float* gat3_as  = (const float*)d_in[19];
  const float* gat3_ad  = (const float*)d_in[20];
  const float* gat3_b   = (const float*)d_in[21];
  const float* dec_w1   = (const float*)d_in[22];
  const float* dec_b1   = (const float*)d_in[23];
  const float* dec_g1   = (const float*)d_in[24];
  const float* dec_bt1  = (const float*)d_in[25];
  const float* dec_w2   = (const float*)d_in[26];
  const float* dec_b2   = (const float*)d_in[27];
  const float* dec_g2   = (const float*)d_in[28];
  const float* dec_bt2  = (const float*)d_in[29];
  const float* dec_w3   = (const float*)d_in[30];
  const float* dec_b3   = (const float*)d_in[31];
  const float* clu_w1   = (const float*)d_in[32];
  const float* clu_b1   = (const float*)d_in[33];
  const float* clu_g1   = (const float*)d_in[34];
  const float* clu_bt1  = (const float*)d_in[35];
  const float* clu_w2   = (const float*)d_in[36];
  const float* clu_b2   = (const float*)d_in[37];

  // Shapes derived from input sizes (reference: N=50000, D=1000, ...).
  const int FH1 = in_sizes[3];              // 256
  const int FH2 = in_sizes[7];              // 128
  const int Dd  = in_sizes[2] / FH1;        // 1000
  const int Nn  = in_sizes[0] / Dd;         // 50000
  const int E   = in_sizes[1] / 2;          // 800000
  const int Etot = E + Nn;                  // self-loops appended
  const int H    = 3;
  const int HGH1 = in_sizes[13];            // 192
  const int GH1  = HGH1 / H;                // 64
  const int GH2  = in_sizes[17];            // 30
  const int NC   = in_sizes[37];            // 10

  float* ws = (float*)d_ws;
  size_t o = 0;
  float*    h1  = ws + o; o += (size_t)Nn * FH1;        // 12.8M  (reused as d2)
  float*    h2  = ws + o; o += (size_t)Nn * FH2;        //  6.4M  (reused as d1)
  float*    hh1 = ws + o; o += (size_t)Nn * HGH1;       //  9.6M  (reused as c1)
  float*    h3  = ws + o; o += (size_t)Nn * HGH1;       //  9.6M
  float*    hh2 = ws + o; o += (size_t)Nn * GH2;
  float*    hh3 = ws + o; o += (size_t)Nn * GH2;
  float*    ss1 = ws + o; o += (size_t)Nn * H;
  float*    sd1 = ws + o; o += (size_t)Nn * H;
  unsigned* m1  = (unsigned*)(ws + o); o += (size_t)Nn * H;
  float*    dn1 = ws + o; o += (size_t)Nn * H;
  float*    e1  = ws + o; o += (size_t)Etot * H;
  float*    ss2 = ws + o; o += Nn;
  float*    sd2 = ws + o; o += Nn;
  unsigned* m2  = (unsigned*)(ws + o); o += Nn;
  float*    dn2 = ws + o; o += Nn;
  float*    e2  = ws + o; o += Etot;
  float*    ss3 = ws + o; o += Nn;
  float*    sd3 = ws + o; o += Nn;
  unsigned* m3  = (unsigned*)(ws + o); o += Nn;
  float*    dn3 = ws + o; o += Nn;
  float*    e3  = ws + o; o += Etot;
  float*    bmean = ws + o; o += 512;
  float*    bvar  = ws + o; o += 512;
  float* d1 = h2;   // decoder hidden 1 reuses h2 (free after hh1 computed)
  float* d2 = h1;   // decoder hidden 2 reuses h1
  float* c1 = hh1;  // cluster hidden reuses hh1 (free after h3 computed)

  float* out     = (float*)d_out;
  float* o_pred  = out;
  float* o_xrec  = out + (size_t)Nn * NC;
  float* o_z     = o_xrec + (size_t)Nn * Dd;
  float* o_mu    = o_z + (size_t)Nn * GH2;
  float* o_lv    = o_mu + (size_t)Nn * GH2;

  const unsigned ORD_NEG_INF = 0x007FFFFFu;  // f2o(-inf)

  auto gemm = [&](const float* A, const float* B, const float* bias, float* C,
                  int M, int Ncol, int K) {
    dim3 g((Ncol + GEMM_BN - 1) / GEMM_BN, (M + GEMM_BM - 1) / GEMM_BM);
    gemm_wmma_f16<<<g, dim3(256), 0, stream>>>(A, B, bias, C, M, Ncol, K);
  };
  auto bn = [&](float* X, const float* g_, const float* bt, int C) {
    bn_stats<<<dim3(C), dim3(256), 0, stream>>>(X, bmean, bvar, Nn, C);
    long tot = (long)Nn * C;
    bn_elu<<<dim3((unsigned)((tot + 255) / 256)), dim3(256), 0, stream>>>(
        X, bmean, bvar, g_, bt, Nn, C);
  };
  auto gat = [&](const float* hh, const float* as, const float* ad,
                 float* ss, float* sd, unsigned* mo, float* dn, float* es,
                 float* aggout, int Hh, int Cc) {
    int NH = Nn * Hh;
    gat_scores<<<dim3((NH + 255) / 256), dim3(256), 0, stream>>>(
        hh, as, ad, ss, sd, Nn, Hh, Cc);
    fill_u32<<<dim3((NH + 255) / 256), dim3(256), 0, stream>>>(mo, ORD_NEG_INF, NH);
    fill_f32<<<dim3((NH + 255) / 256), dim3(256), 0, stream>>>(dn, 0.0f, NH);
    long te = (long)Etot * Hh;
    edge_max<<<dim3((unsigned)((te + 255) / 256)), dim3(256), 0, stream>>>(
        ei, E, Etot, ss, sd, es, mo, Hh);
    edge_expsum<<<dim3((unsigned)((te + 255) / 256)), dim3(256), 0, stream>>>(
        ei, E, Etot, es, mo, dn, Hh);
    long ta = (long)Etot * Hh * Cc;
    edge_agg<<<dim3((unsigned)((ta + 255) / 256)), dim3(256), 0, stream>>>(
        ei, E, Etot, hh, es, dn, aggout, Hh, Cc);
  };
  auto zero = [&](float* p, long n) {
    fill_f32<<<dim3((unsigned)((n + 255) / 256)), dim3(256), 0, stream>>>(p, 0.0f, n);
  };

  // ---- Encoder ----
  gemm(x, enc_w1, enc_b1, h1, Nn, FH1, Dd);
  bn(h1, enc_g1, enc_bt1, FH1);
  gemm(h1, enc_w2, enc_b2, h2, Nn, FH2, FH1);
  bn(h2, enc_g2, enc_bt2, FH2);

  // ---- GAT layer 1 (3 heads, concat, ReLU) ----
  gemm(h2, gat1_w, nullptr, hh1, Nn, HGH1, FH2);
  zero(h3, (long)Nn * HGH1);
  gat(hh1, gat1_as, gat1_ad, ss1, sd1, m1, dn1, e1, h3, H, GH1);
  bias_act<<<dim3((unsigned)(((long)Nn * HGH1 + 255) / 256)), dim3(256), 0, stream>>>(
      h3, gat1_b, Nn, HGH1, 1, nullptr);

  // ---- GAT layer 2 -> mu (and z = mu) ----
  gemm(h3, gat2_w, nullptr, hh2, Nn, GH2, HGH1);
  zero(o_mu, (long)Nn * GH2);
  gat(hh2, gat2_as, gat2_ad, ss2, sd2, m2, dn2, e2, o_mu, 1, GH2);
  bias_act<<<dim3((unsigned)(((long)Nn * GH2 + 255) / 256)), dim3(256), 0, stream>>>(
      o_mu, gat2_b, Nn, GH2, 0, o_z);

  // ---- GAT layer 3 -> logvar ----
  gemm(h3, gat3_w, nullptr, hh3, Nn, GH2, HGH1);
  zero(o_lv, (long)Nn * GH2);
  gat(hh3, gat3_as, gat3_ad, ss3, sd3, m3, dn3, e3, o_lv, 1, GH2);
  bias_act<<<dim3((unsigned)(((long)Nn * GH2 + 255) / 256)), dim3(256), 0, stream>>>(
      o_lv, gat3_b, Nn, GH2, 0, nullptr);

  // ---- Decoder ----
  gemm(o_z, dec_w1, dec_b1, d1, Nn, FH2, GH2);
  bn(d1, dec_g1, dec_bt1, FH2);
  gemm(d1, dec_w2, dec_b2, d2, Nn, FH1, FH2);
  bn(d2, dec_g2, dec_bt2, FH1);
  gemm(d2, dec_w3, dec_b3, o_xrec, Nn, Dd, FH1);

  // ---- Cluster head ----
  gemm(o_z, clu_w1, clu_b1, c1, Nn, FH2, GH2);
  bn(c1, clu_g1, clu_bt1, FH2);
  cluster_head<<<dim3((Nn + 255) / 256), dim3(256), 0, stream>>>(
      c1, clu_w2, clu_b2, o_pred, Nn, FH2, NC);
}
